// GCRUCell_38147899523553
// MI455X (gfx1250) — compile-verified, matrix operand order checked by reading, not run on hardware
//
#include <hip/hip_runtime.h>
#include <hip/hip_bf16.h>

// ---------------------------------------------------------------------------
// GCRU cell for MI455X (gfx1250): f16 WMMA GEMMs + L2-resident edge scatter.
// ---------------------------------------------------------------------------

typedef __attribute__((ext_vector_type(16))) _Float16 v16h;
typedef __attribute__((ext_vector_type(8)))  float    v8f;
typedef __attribute__((ext_vector_type(4)))  int      v4i;

#define NN      50000
#define DH      128
#define DCAT    256
#define NE      800000
#define MTILES  (NN / 16)      // 3125, exact

#define WMMA_F16(A, B, C) \
    __builtin_amdgcn_wmma_f32_16x16x32_f16(false, (A), false, (B), (short)0, (C), false, false)

// Load 16 packed f16 (8 VGPRs) from two 16-byte chunks.
__device__ __forceinline__ v16h load_frag(const _Float16* p0, const _Float16* p1) {
    v16h r;
    reinterpret_cast<v4i*>(&r)[0] = *reinterpret_cast<const v4i*>(p0);
    reinterpret_cast<v4i*>(&r)[1] = *reinterpret_cast<const v4i*>(p1);
    return r;
}

__device__ __forceinline__ float sigmoid_f(float x) {
    return 1.0f / (1.0f + __expf(-x));
}

// ---------------------------------------------------------------------------
// Pre-swizzle a row-major [K, 128] f32 weight into CDNA5 WMMA B-fragments:
// frag[ktile][ntile][lane][16 f16], lane 0-15 = N col, lane>>4 selects K-half.
// f16 halves j=0..7 -> K = k0+j ; j=8..15 -> K = k0+16+(j-8)  (ISA 7.12.2).
// ---------------------------------------------------------------------------
__global__ void __launch_bounds__(256) prep_weight(const float* __restrict__ W,
                                                   _Float16* __restrict__ frag,
                                                   int K) {
    int idx   = blockIdx.x * 256 + threadIdx.x;
    int total = (K / 32) * 8 * 32 * 16;
    if (idx >= total) return;
    int j     = idx & 15;
    int lane  = (idx >> 4) & 31;
    int ntile = (idx >> 9) & 7;
    int ktile = idx >> 12;
    int n     = ntile * 16 + (lane & 15);
    int khalf = lane >> 4;
    int k     = ktile * 32 + khalf * 8 + ((j < 8) ? j : 16 + (j - 8));
    frag[idx] = (_Float16)W[(size_t)k * DH + n];
}

__global__ void __launch_bounds__(256) f32_to_f16(const float* __restrict__ a,
                                                  _Float16* __restrict__ b, int n) {
    int i = blockIdx.x * 256 + threadIdx.x;
    if (i < n) b[i] = (_Float16)a[i];
}

__global__ void __launch_bounds__(256) zero_f32(float* __restrict__ p, int n) {
    int i = blockIdx.x * 256 + threadIdx.x;
    if (i < n) p[i] = 0.0f;
}

// ---------------------------------------------------------------------------
// Fused gate GEMM:  r = sig(x@Wxr + b_xr + h@Whr),  z = sig(x@Wxz + b_xz + h@Whz)
// One wave per 16x16 tile; block = 8 waves = 8 N-tiles; grid = 3125 M-tiles.
// Writes z (f32), r*h (f32 for scatter, f16 for the W_r GEMM).
// ---------------------------------------------------------------------------
__global__ void __launch_bounds__(256) gates_kernel(
    const _Float16* __restrict__ x16,  const _Float16* __restrict__ h16,
    const float* __restrict__ h_prev,
    const float* __restrict__ b_xr,    const float* __restrict__ b_xz,
    const _Float16* __restrict__ fWxr, const _Float16* __restrict__ fWhr,
    const _Float16* __restrict__ fWxz, const _Float16* __restrict__ fWhz,
    float* __restrict__ z_ws, float* __restrict__ rh32, _Float16* __restrict__ rh16)
{
    const int lane  = threadIdx.x & 31;
    const int ntile = threadIdx.x >> 5;
    const int mtile = blockIdx.x;
    const int col   = ntile * 16 + (lane & 15);
    const int khalf = lane >> 4;
    const int arow  = mtile * 16 + (lane & 15);

    v8f accr, accz;
    const float br = b_xr[col], bz = b_xz[col];
#pragma unroll
    for (int v = 0; v < 8; ++v) { accr[v] = br; accz[v] = bz; }

    const _Float16* xb = x16 + (size_t)arow * DH;
    const _Float16* hb = h16 + (size_t)arow * DH;

#pragma unroll
    for (int kt = 0; kt < 4; ++kt) {
        const int ko = kt * 32 + khalf * 8;
        v16h ax = load_frag(xb + ko, xb + ko + 16);
        v16h ah = load_frag(hb + ko, hb + ko + 16);
        const size_t off = ((size_t)(kt * 8 + ntile) * 32 + lane) * 16;
        v16h bxr = load_frag(fWxr + off, fWxr + off + 8);
        v16h bhr = load_frag(fWhr + off, fWhr + off + 8);
        v16h bxz = load_frag(fWxz + off, fWxz + off + 8);
        v16h bhz = load_frag(fWhz + off, fWhz + off + 8);
        accr = WMMA_F16(ax, bxr, accr);
        accr = WMMA_F16(ah, bhr, accr);
        accz = WMMA_F16(ax, bxz, accz);
        accz = WMMA_F16(ah, bhz, accz);
    }

    // Epilogue: C layout — lane 0-15: M = v; lane 16-31: M = v + 8.
    const int mbase = mtile * 16 + ((lane & 16) ? 8 : 0);
#pragma unroll
    for (int v = 0; v < 8; ++v) {
        const size_t idx = (size_t)(mbase + v) * DH + col;
        const float rv = sigmoid_f(accr[v]);
        const float zv = sigmoid_f(accz[v]);
        const float h  = h_prev[idx];
        const float rh = rv * h;
        z_ws[idx] = zv;
        rh32[idx] = rh;
        rh16[idx] = (_Float16)rh;
    }
}

// ---------------------------------------------------------------------------
// Edge scatter: agg[dst] += concat(x, r*h)[src]; cnt[dst] += 1.
// One wave per edge; 8 edges per 256-thread block; coalesced loads + f32 atomics.
// ---------------------------------------------------------------------------
__global__ void __launch_bounds__(256) scatter_kernel(
    const int* __restrict__ ei, const float* __restrict__ x,
    const float* __restrict__ rh32,
    float* __restrict__ agg, float* __restrict__ cnt)
{
    const int lane = threadIdx.x & 31;
    const int wave = threadIdx.x >> 5;
    const int e    = blockIdx.x * 8 + wave;
    const int s    = ei[e];        // src
    const int d    = ei[NE + e];   // dst
    const float* xs = x    + (size_t)s * DH;
    const float* rs = rh32 + (size_t)s * DH;
    float* ag = agg + (size_t)d * DCAT;
#pragma unroll
    for (int i = 0; i < 4; ++i) {
        const int c = lane + i * 32;
        atomicAdd(&ag[c], xs[c]);
    }
#pragma unroll
    for (int i = 0; i < 4; ++i) {
        const int c = lane + i * 32;
        atomicAdd(&ag[DH + c], rs[c]);
    }
    if (lane == 0) atomicAdd(&cnt[d], 1.0f);
}

// ---------------------------------------------------------------------------
// Candidate GEMM + output blend:
//   n = (agg/cnt) @ W_l + b_l + xc @ W_r ;  out = (1-z)*n + z*h_prev
// mean-neighbor A-fragments built in registers (f32 scale -> f16).
// ---------------------------------------------------------------------------
__global__ void __launch_bounds__(256) cand_kernel(
    const float* __restrict__ agg, const float* __restrict__ cnt,
    const _Float16* __restrict__ x16, const _Float16* __restrict__ rh16,
    const _Float16* __restrict__ fWl, const _Float16* __restrict__ fWr,
    const float* __restrict__ b_l,
    const float* __restrict__ z_ws, const float* __restrict__ h_prev,
    float* __restrict__ out)
{
    const int lane  = threadIdx.x & 31;
    const int ntile = threadIdx.x >> 5;
    const int mtile = blockIdx.x;
    const int col   = ntile * 16 + (lane & 15);
    const int khalf = lane >> 4;
    const int arow  = mtile * 16 + (lane & 15);

    const float inv = 1.0f / fmaxf(cnt[arow], 1.0f);

    v8f acc;
    const float bl = b_l[col];
#pragma unroll
    for (int v = 0; v < 8; ++v) acc[v] = bl;

    const float* ab = agg + (size_t)arow * DCAT;

#pragma unroll
    for (int kt = 0; kt < 8; ++kt) {
        const int ko = kt * 32 + khalf * 8;
        // mean-neighbor A fragment, built in registers
        v16h am;
#pragma unroll
        for (int j = 0; j < 8; ++j) {
            am[j]     = (_Float16)(ab[ko + j]      * inv);
            am[8 + j] = (_Float16)(ab[ko + 16 + j] * inv);
        }
        // xc = concat(x, r*h): ktiles 0-3 from x16, 4-7 from rh16
        const _Float16* xcb = (kt < 4)
            ? (x16  + (size_t)arow * DH + kt * 32)
            : (rh16 + (size_t)arow * DH + (kt - 4) * 32);
        v16h ac = load_frag(xcb + khalf * 8, xcb + khalf * 8 + 16);

        const size_t off = ((size_t)(kt * 8 + ntile) * 32 + lane) * 16;
        v16h bwl = load_frag(fWl + off, fWl + off + 8);
        v16h bwr = load_frag(fWr + off, fWr + off + 8);
        acc = WMMA_F16(am, bwl, acc);
        acc = WMMA_F16(ac, bwr, acc);
    }

    const int mbase = mtile * 16 + ((lane & 16) ? 8 : 0);
#pragma unroll
    for (int v = 0; v < 8; ++v) {
        const size_t idx = (size_t)(mbase + v) * DH + col;
        const float zv = z_ws[idx];
        out[idx] = (1.0f - zv) * acc[v] + zv * h_prev[idx];
    }
}

// ---------------------------------------------------------------------------
extern "C" void kernel_launch(void* const* d_in, const int* in_sizes, int n_in,
                              void* d_out, int out_size, void* d_ws, size_t ws_size,
                              hipStream_t stream) {
    const float* x      = (const float*)d_in[0];
    const int*   ei     = (const int*)  d_in[1];
    const float* h_prev = (const float*)d_in[2];
    const float* W_xr   = (const float*)d_in[3];
    const float* b_xr   = (const float*)d_in[4];
    const float* W_hr   = (const float*)d_in[5];
    const float* W_xz   = (const float*)d_in[6];
    const float* b_xz   = (const float*)d_in[7];
    const float* W_hz   = (const float*)d_in[8];
    const float* W_l    = (const float*)d_in[9];
    const float* b_l    = (const float*)d_in[10];
    const float* W_r    = (const float*)d_in[11];
    float* out = (float*)d_out;

    char* ws = (char*)d_ws;
    size_t off = 0;
    auto alloc = [&](size_t bytes) -> char* {
        char* p = ws + off;
        off += (bytes + 255) & ~(size_t)255;
        return p;
    };
    _Float16* x16  = (_Float16*)alloc((size_t)NN * DH * 2);
    _Float16* h16  = (_Float16*)alloc((size_t)NN * DH * 2);
    _Float16* rh16 = (_Float16*)alloc((size_t)NN * DH * 2);
    float*    z_ws = (float*)   alloc((size_t)NN * DH * 4);
    float*    rh32 = (float*)   alloc((size_t)NN * DH * 4);
    float*    agg  = (float*)   alloc((size_t)NN * DCAT * 4);
    float*    cnt  = (float*)   alloc((size_t)NN * 4);
    _Float16* fWxr = (_Float16*)alloc(128 * 128 * 2);
    _Float16* fWhr = (_Float16*)alloc(128 * 128 * 2);
    _Float16* fWxz = (_Float16*)alloc(128 * 128 * 2);
    _Float16* fWhz = (_Float16*)alloc(128 * 128 * 2);
    _Float16* fWl  = (_Float16*)alloc(256 * 128 * 2);
    _Float16* fWr  = (_Float16*)alloc(256 * 128 * 2);

    // 1) pre-swizzle weights into WMMA B-fragment layout
    prep_weight<<<64,  256, 0, stream>>>(W_xr, fWxr, 128);
    prep_weight<<<64,  256, 0, stream>>>(W_hr, fWhr, 128);
    prep_weight<<<64,  256, 0, stream>>>(W_xz, fWxz, 128);
    prep_weight<<<64,  256, 0, stream>>>(W_hz, fWhz, 128);
    prep_weight<<<128, 256, 0, stream>>>(W_l,  fWl,  256);
    prep_weight<<<128, 256, 0, stream>>>(W_r,  fWr,  256);

    // 2) f16 copies of activations; zero the accumulation buffers
    const int nact = NN * DH;
    f32_to_f16<<<(nact + 255) / 256, 256, 0, stream>>>(x,      x16, nact);
    f32_to_f16<<<(nact + 255) / 256, 256, 0, stream>>>(h_prev, h16, nact);
    zero_f32<<<((NN * DCAT) + 255) / 256, 256, 0, stream>>>(agg, NN * DCAT);
    zero_f32<<<(NN + 255) / 256, 256, 0, stream>>>(cnt, NN);

    // 3) fused gate GEMMs (r, z) -> z, r*h
    gates_kernel<<<MTILES, 256, 0, stream>>>(x16, h16, h_prev, b_xr, b_xz,
                                             fWxr, fWhr, fWxz, fWhz,
                                             z_ws, rh32, rh16);

    // 4) edge scatter-sum (mean numerator + counts)
    scatter_kernel<<<NE / 8, 256, 0, stream>>>(ei, x, rh32, agg, cnt);

    // 5) candidate GEMM + GRU blend
    cand_kernel<<<MTILES, 256, 0, stream>>>(agg, cnt, x16, rh16, fWl, fWr,
                                            b_l, z_ws, h_prev, out);
}
